// GATEncoder_52862457479530
// MI455X (gfx1250) — compile-verified, matrix operand order checked by reading, not run on hardware
//
#include <hip/hip_runtime.h>

// ---------------------------------------------------------------------------
// CDNA5 / gfx1250 GAT encoder (3 × GATConv + skip), wave32, WMMA bf16 GEMMs.
// ---------------------------------------------------------------------------

typedef __attribute__((ext_vector_type(16))) __bf16 v16bf;
typedef __attribute__((ext_vector_type(8)))  float  v8f;
typedef int v4i_ __attribute__((vector_size(16)));   // matches builtin pointee

#define HEADS 4
#define CDIM  64
#define HC    256   // HEADS*CDIM

// Async global->LDS copy support (gfx1250). Guarded so absence falls back to
// the plain load+ds_store staging path and the compile stays green.
#if defined(__has_builtin)
#if __has_builtin(__builtin_amdgcn_global_load_async_to_lds_b128)
#define HAVE_ASYNC_LDS 1
#endif
#endif

__device__ __forceinline__ void wait_asynccnt0() {
#if defined(__has_builtin)
#if __has_builtin(__builtin_amdgcn_s_wait_asynccnt)
    __builtin_amdgcn_s_wait_asynccnt(0);
    return;
#endif
#endif
    asm volatile("s_wait_asynccnt 0" ::: "memory");
}

#if defined(HAVE_ASYNC_LDS)
__device__ __forceinline__ void async_b128_to_lds(const float* g, float* lds) {
    // Param types per hipcc diagnostic: (int4 AS1*, int4 AS3*, imm, imm).
    __builtin_amdgcn_global_load_async_to_lds_b128(
        (__attribute__((address_space(1))) v4i_*)(g),
        (__attribute__((address_space(3))) v4i_*)(lds),
        0, 0);
}
#endif

// ---------------------------------------------------------------------------
// Pack a f32 [K x 256] weight matrix into the CDNA5 16-bit B-fragment layout:
//   fragment for (ktile t, coltile ct): lane l<16 -> n = ct*16+l, k = t*32+{0..7,16..23}
//                                       lane l>=16 -> n = ct*16+(l-16), k = t*32+{8..15,24..31}
// Packed as P[((t*16+ct)*32+lane)*16 + slot] so each lane loads 32 contiguous bytes.
// ---------------------------------------------------------------------------
__global__ void pack_b_kernel(const float* __restrict__ W, __bf16* __restrict__ P, int K) {
    int idx = blockIdx.x * 256 + threadIdx.x;          // one thread per (t,ct,lane)
    int total = (K / 32) * 16 * 32;
    if (idx >= total) return;
    int lane = idx & 31;
    int ct   = (idx >> 5) & 15;
    int t    = idx >> 9;
    int n  = ct * 16 + (lane & 15);
    int kb = t * 32 + ((lane < 16) ? 0 : 8);
    __bf16* dst = P + (size_t)idx * 16;
#pragma unroll
    for (int s = 0; s < 8; ++s) {
        dst[s]     = (__bf16)W[(size_t)(kb + s) * HC + n];
        dst[s + 8] = (__bf16)W[(size_t)(kb + 16 + s) * HC + n];
    }
}

// ---------------------------------------------------------------------------
// Dual GEMM: Hout = X@W ; Sout = X@SW + (b + Sb).   X: [N x K] f32.
// Block = 256 threads (8 waves). Each block owns a 16-row M strip.
// Waves 0-3 -> Hout coltiles, waves 4-7 -> Sout coltiles (4 x 16 cols each).
// X strip staged to LDS via async copy (ASYNCcnt) when available, padded
// stride 260 to dodge bank conflicts; converted to bf16 per-lane into the
// documented A-fragment layout; WMMA f32 <- bf16.
// ---------------------------------------------------------------------------
#define XS_STRIDE 260
__global__ void gemm_dual_wmma(const float* __restrict__ X,
                               const __bf16* __restrict__ PW,
                               const __bf16* __restrict__ PS,
                               const float* __restrict__ bias_b,
                               const float* __restrict__ bias_sb,
                               float* __restrict__ Hout,
                               float* __restrict__ Sout,
                               int Nn, int K) {
    __shared__ float xs[16 * XS_STRIDE];               // up to 16x256 f32 strip

    const int tid   = threadIdx.x;
    const int lane  = tid & 31;
    const int wave  = tid >> 5;
    const int mbase = blockIdx.x * 16;

    // Stage 16 consecutive rows of X (rows are contiguous in memory).
    const float* gsrc = X + (size_t)mbase * K;
#if defined(HAVE_ASYNC_LDS)
    {
        const int cpr    = K >> 2;                     // b128 chunks per row
        const int chunks = 16 * cpr;
        for (int i = tid; i < chunks; i += 256) {
            int r = i / cpr;
            // Clamp OOB rows to a valid address (their outputs are never stored).
            const float* g = ((mbase + r) < Nn) ? (gsrc + (size_t)i * 4) : X;
            int c4 = i - r * cpr;
            async_b128_to_lds(g, &xs[r * XS_STRIDE + c4 * 4]);
        }
        wait_asynccnt0();
    }
#else
    for (int i = tid; i < 16 * K; i += 256) {
        int r = i / K, c = i - r * K;
        float v = ((mbase + r) < Nn) ? gsrc[i] : 0.0f;
        xs[r * XS_STRIDE + c] = v;
    }
#endif
    __syncthreads();

    const bool   isSkip = (wave >= 4);
    const int    ct0    = (wave & 3) * 4;
    const __bf16* P     = isSkip ? PS : PW;
    float*       Out    = isSkip ? Sout : Hout;

    v8f acc[4] = {};

    const int m    = lane & 15;
    const int kofs = (lane < 16) ? 0 : 8;

    for (int t = 0; t < (K >> 5); ++t) {
        // A fragment: lane m row, K slots {kb..kb+7, kb+16..kb+23}
        v16bf afrag;
        const float* row = xs + m * XS_STRIDE + t * 32 + kofs;
#pragma unroll
        for (int s = 0; s < 8; ++s) {
            afrag[s]     = (__bf16)row[s];
            afrag[s + 8] = (__bf16)row[16 + s];
        }
#pragma unroll
        for (int j = 0; j < 4; ++j) {
            const v16bf bfrag =
                *(const v16bf*)(P + ((((size_t)t * 16 + (ct0 + j)) * 32 + lane) << 4));
            acc[j] = __builtin_amdgcn_wmma_f32_16x16x32_bf16(
                false, afrag, false, bfrag, (short)0, acc[j], false, false);
        }
    }

    // Epilogue: C/D layout -> lane<16: M=i, N=lane ; lane>=16: M=i+8, N=lane-16
    const int mofs = (lane < 16) ? 0 : 8;
#pragma unroll
    for (int j = 0; j < 4; ++j) {
        const int ncol = (ct0 + j) * 16 + m;
        const float bv = isSkip ? (bias_b[ncol] + bias_sb[ncol]) : 0.0f;
#pragma unroll
        for (int i = 0; i < 8; ++i) {
            int grow = mbase + mofs + i;
            if (grow < Nn) Out[(size_t)grow * HC + ncol] = acc[j][i] + bv;
        }
    }
}

// ---------------------------------------------------------------------------
// s[n,h] = <h[n,h,:], a_src[h,:]> ; d[n,h] = <h[n,h,:], a_dst[h,:]>
// ---------------------------------------------------------------------------
__global__ void attn_sd_kernel(const float* __restrict__ h,
                               const float* __restrict__ a_src,
                               const float* __restrict__ a_dst,
                               float* __restrict__ s, float* __restrict__ d, int Nn) {
    int idx = blockIdx.x * 256 + threadIdx.x;
    if (idx >= Nn * HEADS) return;
    int head = idx & 3, n = idx >> 2;
    const float* hp = h + (size_t)n * HC + head * CDIM;
    const float* as = a_src + head * CDIM;
    const float* ad = a_dst + head * CDIM;
    float ss = 0.f, dd = 0.f;
#pragma unroll 8
    for (int c = 0; c < CDIM; ++c) {
        float v = hp[c];
        ss += v * as[c];
        dd += v * ad[c];
    }
    s[idx] = ss;
    d[idx] = dd;
}

__global__ void fill_zero_kernel(unsigned int* __restrict__ p, size_t n) {
    size_t i = (size_t)blockIdx.x * 256 + threadIdx.x;
    if (i < n) p[i] = 0u;
}

__device__ __forceinline__ float leaky02(float v) { return v > 0.f ? v : 0.2f * v; }
__device__ __forceinline__ unsigned f2key(float f) {
    unsigned u = __float_as_uint(f);
    return (u & 0x80000000u) ? ~u : (u | 0x80000000u);   // monotonic total order
}
__device__ __forceinline__ float key2f(unsigned key) {
    unsigned u = (key & 0x80000000u) ? (key ^ 0x80000000u) : ~key;
    return __uint_as_float(u);
}
__device__ __forceinline__ void edge_ends(const int* ei, int E, int e, int& src, int& dst) {
    if (e < E) { src = ei[e]; dst = ei[E + e]; }
    else       { src = dst = e - E; }                    // appended self-loops
}

// Pass 1: segment max (keyed atomicMax)
__global__ void edge_max_kernel(const int* __restrict__ ei, int E, int Nn,
                                const float* __restrict__ s, const float* __restrict__ d,
                                unsigned int* __restrict__ mkey) {
    int e = blockIdx.x * 256 + threadIdx.x;
    if (e >= E + Nn) return;
    int src, dst; edge_ends(ei, E, e, src, dst);
#pragma unroll
    for (int hh = 0; hh < HEADS; ++hh) {
        float v = leaky02(s[src * HEADS + hh] + d[dst * HEADS + hh]);
        atomicMax(&mkey[dst * HEADS + hh], f2key(v));
    }
}

// Pass 2: segment sum of exp(e - m)
__global__ void edge_den_kernel(const int* __restrict__ ei, int E, int Nn,
                                const float* __restrict__ s, const float* __restrict__ d,
                                const unsigned int* __restrict__ mkey,
                                float* __restrict__ denom) {
    int e = blockIdx.x * 256 + threadIdx.x;
    if (e >= E + Nn) return;
    int src, dst; edge_ends(ei, E, e, src, dst);
#pragma unroll
    for (int hh = 0; hh < HEADS; ++hh) {
        float v = leaky02(s[src * HEADS + hh] + d[dst * HEADS + hh]);
        float m = key2f(mkey[dst * HEADS + hh]);
        atomicAdd(&denom[dst * HEADS + hh], __expf(v - m));
    }
}

// Pass 3: wave-per-edge scatter:  aggr[dst] += alpha * h[src]  (256 f32 / edge)
__global__ void edge_aggr_kernel(const int* __restrict__ ei, int E, int Nn,
                                 const float* __restrict__ s, const float* __restrict__ d,
                                 const unsigned int* __restrict__ mkey,
                                 const float* __restrict__ denom,
                                 const float* __restrict__ h,
                                 float* __restrict__ aggr) {
    int gid  = blockIdx.x * 256 + threadIdx.x;
    int lane = gid & 31;
    int e    = gid >> 5;
    if (e >= E + Nn) return;
    int src, dst; edge_ends(ei, E, e, src, dst);

    const float* hp = h + (size_t)src * HC + lane * 8;
    __builtin_prefetch(hp, 0, 1);                       // global_prefetch_b8 (L2 warm)

    int head = lane >> 3;                               // lane*8 / 64
    float v  = leaky02(s[src * HEADS + head] + d[dst * HEADS + head]);
    float m  = key2f(mkey[dst * HEADS + head]);
    float alpha = __expf(v - m) / (denom[dst * HEADS + head] + 1e-16f);

    float4 h0 = ((const float4*)hp)[0];
    float4 h1 = ((const float4*)hp)[1];
    float* op = aggr + (size_t)dst * HC + lane * 8;
    atomicAdd(op + 0, alpha * h0.x);
    atomicAdd(op + 1, alpha * h0.y);
    atomicAdd(op + 2, alpha * h0.z);
    atomicAdd(op + 3, alpha * h0.w);
    atomicAdd(op + 4, alpha * h1.x);
    atomicAdd(op + 5, alpha * h1.y);
    atomicAdd(op + 6, alpha * h1.z);
    atomicAdd(op + 7, alpha * h1.w);
}

// x_next = elu(aggr + skip)   (skip already contains x@SW + b + Sb)
__global__ void finalize_kernel(const float* __restrict__ aggr,
                                const float* __restrict__ skip,
                                float* __restrict__ out, size_t total) {
    size_t i = (size_t)blockIdx.x * 256 + threadIdx.x;
    if (i >= total) return;
    float v = aggr[i] + skip[i];
    out[i] = v > 0.f ? v : (__expf(v) - 1.f);
}

// ---------------------------------------------------------------------------
// Host-side launcher (graph-capture safe: stream-only, no alloc/sync).
// Input order: 0:x 1:edge_index then per layer {W, a_src, a_dst, b, SW, Sb}.
// ---------------------------------------------------------------------------
extern "C" void kernel_launch(void* const* d_in, const int* in_sizes, int n_in,
                              void* d_out, int out_size, void* d_ws, size_t ws_size,
                              hipStream_t stream) {
    const int IN = 128;
    const int N  = in_sizes[0] / IN;
    const int E  = in_sizes[1] / 2;
    const int ET = E + N;

    const float* x0 = (const float*)d_in[0];
    const int*   ei = (const int*)d_in[1];   // edge_index (int32: JAX x64 disabled)

    // Carve workspace (256B aligned chunks).
    char* w = (char*)d_ws;
    auto carve = [&](size_t bytes) -> void* {
        void* p = (void*)w;
        w += (bytes + 255) & ~(size_t)255;
        return p;
    };
    const size_t nodef = (size_t)N * HC;
    float*        xcur   = (float*)carve(nodef * 4);
    float*        hbuf   = (float*)carve(nodef * 4);
    float*        skip   = (float*)carve(nodef * 4);
    float*        aggr   = (float*)carve(nodef * 4);
    float*        sbuf   = (float*)carve((size_t)N * HEADS * 4);
    float*        dbuf   = (float*)carve((size_t)N * HEADS * 4);
    unsigned int* mkey   = (unsigned int*)carve((size_t)N * HEADS * 4);
    float*        denom  = (float*)carve((size_t)N * HEADS * 4);
    __bf16*       packW  = (__bf16*)carve((size_t)HC * HC * 2);
    __bf16*       packSW = (__bf16*)carve((size_t)HC * HC * 2);

    const int mtiles = (N + 15) / 16;

    for (int l = 0; l < 3; ++l) {
        const int K = (l == 0) ? IN : HC;
        const float* W  = (const float*)d_in[2 + 6 * l + 0];
        const float* as = (const float*)d_in[2 + 6 * l + 1];
        const float* ad = (const float*)d_in[2 + 6 * l + 2];
        const float* b  = (const float*)d_in[2 + 6 * l + 3];
        const float* SW = (const float*)d_in[2 + 6 * l + 4];
        const float* Sb = (const float*)d_in[2 + 6 * l + 5];
        const float* xin = (l == 0) ? x0 : xcur;
        float* xout = (l == 2) ? (float*)d_out : xcur;

        // Pack weights into WMMA B-fragment layout (bf16).
        int packThreads = (K / 32) * 16 * 32;
        hipLaunchKernelGGL(pack_b_kernel, dim3((packThreads + 255) / 256), dim3(256),
                           0, stream, W, packW, K);
        hipLaunchKernelGGL(pack_b_kernel, dim3((packThreads + 255) / 256), dim3(256),
                           0, stream, SW, packSW, K);

        // Dual GEMM: h = x@W ; skip = x@SW + b + Sb
        hipLaunchKernelGGL(gemm_dual_wmma, dim3(mtiles), dim3(256), 0, stream,
                           xin, packW, packSW, b, Sb, hbuf, skip, N, K);

        // Attention logits per node.
        hipLaunchKernelGGL(attn_sd_kernel, dim3((N * HEADS + 255) / 256), dim3(256),
                           0, stream, hbuf, as, ad, sbuf, dbuf, N);

        // Zero segment state + accumulator.
        hipLaunchKernelGGL(fill_zero_kernel, dim3((unsigned)((N * HEADS + 255) / 256)),
                           dim3(256), 0, stream, mkey, (size_t)N * HEADS);
        hipLaunchKernelGGL(fill_zero_kernel, dim3((unsigned)((N * HEADS + 255) / 256)),
                           dim3(256), 0, stream, (unsigned int*)denom, (size_t)N * HEADS);
        hipLaunchKernelGGL(fill_zero_kernel, dim3((unsigned)((nodef + 255) / 256)),
                           dim3(256), 0, stream, (unsigned int*)aggr, nodef);

        // Segment softmax (3 passes) + aggregation.
        hipLaunchKernelGGL(edge_max_kernel, dim3((ET + 255) / 256), dim3(256), 0, stream,
                           ei, E, N, sbuf, dbuf, mkey);
        hipLaunchKernelGGL(edge_den_kernel, dim3((ET + 255) / 256), dim3(256), 0, stream,
                           ei, E, N, sbuf, dbuf, mkey, denom);
        hipLaunchKernelGGL(edge_aggr_kernel,
                           dim3((unsigned)(((size_t)ET * 32 + 255) / 256)), dim3(256),
                           0, stream, ei, E, N, sbuf, dbuf, mkey, denom, hbuf, aggr);

        // x_next = elu(aggr + skip)
        hipLaunchKernelGGL(finalize_kernel, dim3((unsigned)((nodef + 255) / 256)),
                           dim3(256), 0, stream, aggr, skip, xout, nodef);
    }
    (void)n_in; (void)out_size; (void)ws_size;
}
